// Multi_Headed_Attention_51333449122098
// MI455X (gfx1250) — matrix operand order repaired and checked by reading, not compiled, not run on hardware
//
#include <hip/hip_runtime.h>
#include <hip/hip_bf16.h>

// ---------------------------------------------------------------------------
// Multi-headed attention for MI455X (gfx1250, wave32, WMMA, async-to-LDS).
// Pass 0: convert x, Wq, Wk, Wv to bf16 (one cheap streaming pass).
// Pass 1: QKV projection GEMM; block tile M=256 N=64, 32 rows/wave so each
//         B fragment feeds 2 WMMAs; double-buffered
//         GLOBAL_LOAD_ASYNC_TO_LDS_B128 staging; grouped loads -> grouped
//         v_wmma_f32_16x16x32_bf16 batches.
// Pass 2: flash attention; K/V^T tiles async-staged into LDS shared by all
//         8 waves, online softmax via wave32 xor-shuffles, WMMA for QK^T+PV.
// Workspace: xb 8MB + 3 W 6MB + Q/K/Vt 24MB = 38 MB in d_ws.
// ---------------------------------------------------------------------------

#define BATCH 2
#define SEQ   2048
#define EMBED 1024
#define NH    16
#define DKEY  64
#define NEGV  (-1.0e9f)

typedef __attribute__((ext_vector_type(16))) __bf16 v16bf;
typedef __attribute__((ext_vector_type(8)))  float  v8f;

struct Bf16Frag { uint4 lo, hi; };

__device__ __forceinline__ v16bf frag_from(uint4 a, uint4 b) {
  Bf16Frag t{a, b};
  return __builtin_bit_cast(v16bf, t);
}

__device__ __forceinline__ uint4 ld4(const unsigned short* p) {
  return *(const uint4*)p;
}

__device__ __forceinline__ unsigned f2bf(float f) {
  unsigned u = __float_as_uint(f);
  u += 0x7FFFu + ((u >> 16) & 1u);       // round-to-nearest-even
  return u >> 16;
}

__device__ __forceinline__ v8f wmma_bf16(v16bf a, v16bf b, v8f c) {
  return __builtin_amdgcn_wmma_f32_16x16x32_bf16(
      false, a, false, b, (short)0, c, false, false);
}

// CDNA5 async copy: 16 bytes per lane, global -> LDS, tracked by ASYNCcnt.
__device__ __forceinline__ void async_copy16(const unsigned short* lds_ptr,
                                             const unsigned short* gptr) {
  const unsigned lds_off = (unsigned)(size_t)lds_ptr;  // low 32b = LDS offset
  asm volatile("global_load_async_to_lds_b128 %0, %1, off"
               :: "v"(lds_off), "v"(gptr)
               : "memory");
}

__device__ __forceinline__ void wait_async0() {
  asm volatile("s_wait_asynccnt 0" ::: "memory");
}

// ---------------------------------------------------------------------------
// Pass 0: f32 -> bf16 streaming convert (8 elements / thread).
// ---------------------------------------------------------------------------
__global__ __launch_bounds__(256)
void cvt_bf16_kernel(const float* __restrict__ src,
                     unsigned short* __restrict__ dst, int n8) {
  const int i = blockIdx.x * 256 + threadIdx.x;
  if (i < n8) {
    const float4 f0 = ((const float4*)src)[2 * i];
    const float4 f1 = ((const float4*)src)[2 * i + 1];
    uint4 o;
    o.x = f2bf(f0.x) | (f2bf(f0.y) << 16);
    o.y = f2bf(f0.z) | (f2bf(f0.w) << 16);
    o.z = f2bf(f1.x) | (f2bf(f1.y) << 16);
    o.w = f2bf(f1.z) | (f2bf(f1.w) << 16);
    ((uint4*)dst)[i] = o;
  }
}

// ---------------------------------------------------------------------------
// Pass 1: QKV projection  y = x @ W^T + bias   (4096 x 1024 x 1024, x3)
// Block: 256 threads (8 waves). Tile: M=256 (32 rows/wave), N=64, K-step 32,
// double-buffered async staging.  grid = (16 M-tiles, 16 N-tiles, 3 mats)
// ---------------------------------------------------------------------------
__global__ __launch_bounds__(256)
void qkv_proj_kernel(const unsigned short* __restrict__ xb,
                     const unsigned short* __restrict__ Wqb,
                     const unsigned short* __restrict__ Wkb,
                     const unsigned short* __restrict__ Wvb,
                     const float* __restrict__ bq,
                     const float* __restrict__ bk,
                     const float* __restrict__ bv,
                     unsigned short* __restrict__ Qh,
                     unsigned short* __restrict__ Kh,
                     unsigned short* __restrict__ Vt)
{
  const int mt  = blockIdx.x;
  const int nt  = blockIdx.y;
  const int mat = blockIdx.z;                 // 0=Q 1=K 2=V
  const unsigned short* W = (mat == 0) ? Wqb : (mat == 1) ? Wkb : Wvb;
  const float* bias       = (mat == 0) ? bq  : (mat == 1) ? bk  : bv;

  const int tid  = threadIdx.x;
  const int wave = tid >> 5;
  const int lane = tid & 31;
  const int l16  = lane & 15;
  const int lh   = lane >> 4;

  constexpr int RS = 40;                      // LDS row stride (halves)
  __shared__ unsigned short Xs[2][256 * RS];  // 40 KB
  __shared__ unsigned short Wsh[2][64 * RS];  // 10 KB

  const int mbase = mt * 256;
  const int nbase = nt * 64;
  const unsigned short* xsrc = xb + (size_t)mbase * EMBED;
  const unsigned short* wsrc = W  + (size_t)nbase * EMBED;

  auto stage = [&](int buf, int kb) {
    // X tile 256x32 bf16 = 1024 x 16B chunks (4/thread),
    // W tile  64x32 bf16 =  256 x 16B chunks (1/thread)
    for (int c = tid; c < 1024; c += 256) {
      const int r = c >> 2, c8 = (c & 3) * 8;
      async_copy16(&Xs[buf][r * RS + c8], xsrc + r * EMBED + kb + c8);
    }
    {
      const int r = tid >> 2, c8 = (tid & 3) * 8;
      async_copy16(&Wsh[buf][r * RS + c8], wsrc + r * EMBED + kb + c8);
    }
  };

  v8f acc[2][4] = {};
  stage(0, 0);
  for (int it = 0; it < EMBED / 32; ++it) {
    const int buf = it & 1;
    wait_async0();
    __syncthreads();                          // tile visible to all waves
    if (it + 1 < EMBED / 32) stage(buf ^ 1, (it + 1) * 32);

    // ---- grouped fragment loads (one dscnt wait covers the WMMA batch)
    const unsigned short* xr0 = &Xs[buf][(wave * 32 + l16) * RS];
    const unsigned short* xr1 = xr0 + 16 * RS;
    const v16bf a0 = frag_from(ld4(xr0 + lh * 8), ld4(xr0 + 16 + lh * 8));
    const v16bf a1 = frag_from(ld4(xr1 + lh * 8), ld4(xr1 + 16 + lh * 8));
    v16bf b[4];
#pragma unroll
    for (int j = 0; j < 4; ++j) {
      const unsigned short* wr = &Wsh[buf][(j * 16 + l16) * RS + lh * 8];
      b[j] = frag_from(ld4(wr), ld4(wr + 16));
    }
    // ---- 8 WMMAs: each B fragment reused by both A strips
#pragma unroll
    for (int j = 0; j < 4; ++j) {
      acc[0][j] = wmma_bf16(a0, b[j], acc[0][j]);
      acc[1][j] = wmma_bf16(a1, b[j], acc[1][j]);
    }
    __syncthreads();                          // done reading before overwrite
  }

  // Epilogue: bias, fold 1/sqrt(Dk) into Q, scatter
  const float qscale = (mat == 0) ? 0.125f : 1.0f;
#pragma unroll
  for (int s = 0; s < 2; ++s) {
#pragma unroll
    for (int j = 0; j < 4; ++j) {
      const int e = nbase + j * 16 + l16;
      const int h = e >> 6, d = e & 63;
      const float bb = bias[e];
#pragma unroll
      for (int v = 0; v < 8; ++v) {
        const int tok = mbase + wave * 32 + s * 16 + v + lh * 8;
        const int b_ = tok >> 11, s_ = tok & (SEQ - 1);
        const unsigned short hv =
            (unsigned short)f2bf((acc[s][j][v] + bb) * qscale);
        const long long bh = (long long)(b_ * NH + h);
        if (mat == 0)      Qh[(bh * SEQ + s_) * DKEY + d] = hv;
        else if (mat == 1) Kh[(bh * SEQ + s_) * DKEY + d] = hv;
        else               Vt[(bh * DKEY + d) * SEQ + s_] = hv;  // transposed
      }
    }
  }
}

// ---------------------------------------------------------------------------
// Pass 2: flash attention. grid = (B*H = 32, 16 q-tiles of 128).
// K/V^T 32-key tiles async-staged into LDS (shared by 8 waves, double-buf).
// ---------------------------------------------------------------------------
__global__ __launch_bounds__(256)
void attn_kernel(const unsigned short* __restrict__ Qh,
                 const unsigned short* __restrict__ Kh,
                 const unsigned short* __restrict__ Vt,
                 const int* __restrict__ is_masked_p,
                 float* __restrict__ out)
{
  const int bh     = blockIdx.x;
  const int qt     = blockIdx.y;
  const int masked = is_masked_p[0];

  const int tid  = threadIdx.x;
  const int wave = tid >> 5;
  const int lane = tid & 31;
  const int l16  = lane & 15;
  const int lh   = lane >> 4;

  const int b_ = bh >> 4;
  const int h_ = bh & 15;

  const unsigned short* Qb = Qh + (long long)bh * SEQ * DKEY;
  const unsigned short* Kb = Kh + (long long)bh * SEQ * DKEY;
  const unsigned short* Vb = Vt + (long long)bh * DKEY * SEQ;

  constexpr int KRS = 72;  // K tile row stride (32 rows x 64 halves)
  constexpr int VRS = 40;  // V^T tile row stride (64 rows x 32 halves)
  constexpr int PS  = 40;  // P scratch row stride
  __shared__ unsigned short Kts[2][32 * KRS];
  __shared__ unsigned short Vts[2][64 * VRS];
  __shared__ unsigned short Ps[8][16 * PS];
  unsigned short* myP = Ps[wave];

  const int q0 = qt * 128 + wave * 16;

  // Loop-invariant Q A-fragments
  const unsigned short* qrow = Qb + (q0 + l16) * DKEY;
  v16bf qa[2];
#pragma unroll
  for (int dc = 0; dc < 2; ++dc)
    qa[dc] = frag_from(ld4(qrow + dc * 32 + lh * 8),
                       ld4(qrow + dc * 32 + 16 + lh * 8));

  v8f accO[4] = {};
  float mrow[8], lrow[8];
#pragma unroll
  for (int v = 0; v < 8; ++v) { mrow[v] = NEGV; lrow[v] = 0.0f; }

  auto stage = [&](int buf, int kb) {
    // K tile: 32 keys x 64 d = 256 x 16B chunks (1/thread)
    {
      const int r = tid >> 3, c8 = (tid & 7) * 8;
      async_copy16(&Kts[buf][r * KRS + c8], Kb + (kb + r) * DKEY + c8);
    }
    // V^T tile: 128 rows x 16 keys = 256 x 16B chunks (1/thread)
    {
      const int d = tid >> 1, c8 = (tid & 1) * 8;
      async_copy16(&Vts[buf][d * VRS + c8], Vb + (size_t)d * SEQ + kb + c8);
    }
  };

  const int kend_blk = masked ? (qt * 128 + 128) : SEQ;  // uniform in block
  const int kend_w   = masked ? (q0 + 16)        : SEQ;  // uniform in wave

  stage(0, 0);
  for (int kb = 0, it = 0; kb < kend_blk; kb += 32, ++it) {
    const int buf = it & 1;
    wait_async0();
    __syncthreads();
    if (kb + 32 < kend_blk) stage(buf ^ 1, kb + 32);

    if (kb < kend_w) {   // wave-uniform: EXEC stays all-ones for WMMA
      // ---- grouped K-fragment loads, then 4 score WMMAs
      v16bf kf[2][2];
#pragma unroll
      for (int t = 0; t < 2; ++t)
#pragma unroll
        for (int dc = 0; dc < 2; ++dc) {
          const unsigned short* kp =
              &Kts[buf][(t * 16 + l16) * KRS + dc * 32 + lh * 8];
          kf[t][dc] = frag_from(ld4(kp), ld4(kp + 16));
        }
      v8f s0 = {}, s1 = {};
      s0 = wmma_bf16(qa[0], kf[0][0], s0);
      s1 = wmma_bf16(qa[0], kf[1][0], s1);
      s0 = wmma_bf16(qa[1], kf[0][1], s0);
      s1 = wmma_bf16(qa[1], kf[1][1], s1);

      // ---- masking + row max (C layout: row = v + 8*lh, col = lane&15)
      float newm[8];
#pragma unroll
      for (int v = 0; v < 8; ++v) {
        const int q  = q0 + v + lh * 8;
        const int k0 = kb + l16, k1 = kb + 16 + l16;
        float a0 = s0[v], a1 = s1[v];
        if (a0 == 0.0f || (masked && k0 > q)) a0 = NEGV;
        if (a1 == 0.0f || (masked && k1 > q)) a1 = NEGV;
        s0[v] = a0; s1[v] = a1;
        float mx = fmaxf(a0, a1);
        mx = fmaxf(mx, __shfl_xor(mx, 1, 32));
        mx = fmaxf(mx, __shfl_xor(mx, 2, 32));
        mx = fmaxf(mx, __shfl_xor(mx, 4, 32));
        mx = fmaxf(mx, __shfl_xor(mx, 8, 32));
        newm[v] = fmaxf(mrow[v], mx);
      }

      // ---- exp, row sums, rescale, stage P (bf16) in per-wave LDS
#pragma unroll
      for (int v = 0; v < 8; ++v) {
        const float mn = newm[v];
        const float p0 = __expf(s0[v] - mn);
        const float p1 = __expf(s1[v] - mn);
        float rs = p0 + p1;
        rs += __shfl_xor(rs, 1, 32);
        rs += __shfl_xor(rs, 2, 32);
        rs += __shfl_xor(rs, 4, 32);
        rs += __shfl_xor(rs, 8, 32);
        const float sc = __expf(mrow[v] - mn);
        lrow[v] = lrow[v] * sc + rs;
        mrow[v] = mn;
#pragma unroll
        for (int j = 0; j < 4; ++j) accO[j][v] *= sc;
        const int r = v + lh * 8;
        myP[r * PS + l16]      = (unsigned short)f2bf(p0);
        myP[r * PS + 16 + l16] = (unsigned short)f2bf(p1);
      }

      // ---- grouped P/V fragment loads, then 4 PV WMMAs
      const v16bf pa = frag_from(ld4(&myP[l16 * PS + lh * 8]),
                                 ld4(&myP[l16 * PS + 16 + lh * 8]));
      v16bf vb4[4];
#pragma unroll
      for (int j = 0; j < 4; ++j) {
        const unsigned short* vp =
            &Vts[buf][(j * 16 + l16) * VRS + lh * 8];
        vb4[j] = frag_from(ld4(vp), ld4(vp + 16));
      }
#pragma unroll
      for (int j = 0; j < 4; ++j) accO[j] = wmma_bf16(pa, vb4[j], accO[j]);
    }
    __syncthreads();
  }

  // ---- normalize + write f32 output [B, S, H*D]
#pragma unroll
  for (int j = 0; j < 4; ++j) {
#pragma unroll
    for (int v = 0; v < 8; ++v) {
      const int q = q0 + v + lh * 8;
      const int e = h_ * 64 + j * 16 + l16;
      out[((long long)(b_ * SEQ + q)) * EMBED + e] = accO[j][v] / lrow[v];
    }
  }
}

// ---------------------------------------------------------------------------
extern "C" void kernel_launch(void* const* d_in, const int* in_sizes, int n_in,
                              void* d_out, int out_size, void* d_ws, size_t ws_size,
                              hipStream_t stream) {
  const float* x  = (const float*)d_in[0];
  const float* Wq = (const float*)d_in[1];
  const float* bq = (const float*)d_in[2];
  const float* Wk = (const float*)d_in[3];
  const float* bk = (const float*)d_in[4];
  const float* Wv = (const float*)d_in[5];
  const float* bv = (const float*)d_in[6];
  const int*   mk = (const int*)d_in[7];
  float* out = (float*)d_out;

  const size_t nX = (size_t)BATCH * SEQ * EMBED;      // 4M elems
  const size_t nW = (size_t)EMBED * EMBED;            // 1M elems
  const size_t nT = (size_t)BATCH * NH * SEQ * DKEY;  // 4M elems

  unsigned short* xb  = (unsigned short*)d_ws;
  unsigned short* Wqb = xb  + nX;
  unsigned short* Wkb = Wqb + nW;
  unsigned short* Wvb = Wkb + nW;
  unsigned short* Qhp = Wvb + nW;
  unsigned short* Khp = Qhp + nT;
  unsigned short* Vtp = Khp + nT;

  // Pass 0: bf16 conversion
  cvt_bf16_kernel<<<dim3((unsigned)(nX / 8 / 256)), 256, 0, stream>>>(x, xb, (int)(nX / 8));
  cvt_bf16_kernel<<<dim3((unsigned)(nW / 8 / 256)), 256, 0, stream>>>(Wq, Wqb, (int)(nW / 8));
  cvt_bf16_kernel<<<dim3((unsigned)(nW / 8 / 256)), 256, 0, stream>>>(Wk, Wkb, (int)(nW / 8));
  cvt_bf16_kernel<<<dim3((unsigned)(nW / 8 / 256)), 256, 0, stream>>>(Wv, Wvb, (int)(nW / 8));

  // Pass 1: QKV projection
  qkv_proj_kernel<<<dim3(16, 16, 3), 256, 0, stream>>>(
      xb, Wqb, Wkb, Wvb, bq, bk, bv, Qhp, Khp, Vtp);

  // Pass 2: attention
  attn_kernel<<<dim3(32, 16), 256, 0, stream>>>(Qhp, Khp, Vtp, mk, out);
}
